// simpleRQA_46995532153135
// MI455X (gfx1250) — matrix operand order repaired and checked by reading, not compiled, not run on hardware
//
#include <hip/hip_runtime.h>
#include <stdint.h>

typedef __attribute__((ext_vector_type(16))) _Float16 v16h;
typedef __attribute__((ext_vector_type(8)))  _Float16 v8h;
typedef __attribute__((ext_vector_type(4)))  _Float16 v4h;
typedef __attribute__((ext_vector_type(8)))  float    v8f;
typedef __attribute__((ext_vector_type(4)))  float    v4f;

#define N_ROWS 8192
#define DIM    1024
#define PHI    0.9f
#define EPS_N  1e-8f

#define BLK_M 256                   // block rows   (8 waves x 32)
#define BLK_N 128                   // block cols   (each wave covers all 128)
#define KC    32                    // K-chunk in halfs (one WMMA K-step)
#define NCHUNK (DIM / KC)           // 32
#define LDS_STRIDE 40               // 32 halfs + 8 pad -> 80B rows, minimal bank conflicts

// ---------------------------------------------------------------------------
// CDNA5 async global->LDS copy (ASYNCcnt-tracked), ISA 15.18.3 op 98.
// VDST VGPR holds the LDS byte offset; VADDR pair holds the 64-bit address.
// Low 32 bits of a generic pointer to __shared__ are the LDS offset on AMDGPU.
// ---------------------------------------------------------------------------
__device__ __forceinline__ void async_copy_b128(const _Float16* g, _Float16* lds) {
    unsigned l = (unsigned)(uintptr_t)lds;
    asm volatile("global_load_async_to_lds_b128 %0, %1, off"
                 :: "v"(l), "v"(g) : "memory");
}
__device__ __forceinline__ void wait_async0() {
    asm volatile("s_wait_asynccnt 0x0" ::: "memory");
}

// ---------------------------------------------------------------------------
// Kernel 0: zero the output flags (harness poisons d_out with 0xAA).
// ---------------------------------------------------------------------------
__global__ void zero_out_kernel(float* __restrict__ out, int n) {
    int i = blockIdx.x * blockDim.x + threadIdx.x;
    if (i < n) out[i] = 0.0f;
}

// ---------------------------------------------------------------------------
// Kernel 1: L2-normalize each row (clamped at eps) and emit FP16 copies.
// One 256-thread block per row; rows 0..8191 = Ex, 8192..16383 = Ey.
// ---------------------------------------------------------------------------
__global__ __launch_bounds__(256) void normalize_f16_kernel(
    const float* __restrict__ Ex, const float* __restrict__ Ey,
    _Float16* __restrict__ Exh, _Float16* __restrict__ Eyh)
{
    const int row = blockIdx.x;  // 0..2*N_ROWS-1
    const float* src;
    _Float16*    dst;
    if (row < N_ROWS) {
        src = Ex + (size_t)row * DIM;
        dst = Exh + (size_t)row * DIM;
    } else {
        src = Ey + (size_t)(row - N_ROWS) * DIM;
        dst = Eyh + (size_t)(row - N_ROWS) * DIM;
    }

    const int t = threadIdx.x;            // 256 threads * 4 floats = 1024
    v4f x = *(const v4f*)(src + t * 4);
    float ss = x.x * x.x + x.y * x.y + x.z * x.z + x.w * x.w;

    #pragma unroll
    for (int off = 16; off > 0; off >>= 1)
        ss += __shfl_xor(ss, off, 32);

    __shared__ float warp_s[8];
    if ((t & 31) == 0) warp_s[t >> 5] = ss;
    __syncthreads();

    float tot = 0.0f;
    #pragma unroll
    for (int i = 0; i < 8; ++i) tot += warp_s[i];

    const float scale = 1.0f / fmaxf(sqrtf(tot), EPS_N);

    v4h h;
    h.x = (_Float16)(x.x * scale);
    h.y = (_Float16)(x.y * scale);
    h.z = (_Float16)(x.z * scale);
    h.w = (_Float16)(x.w * scale);
    *(v4h*)(dst + t * 4) = h;
}

// ---------------------------------------------------------------------------
// Kernel 2: LDS-staged, double-buffered WMMA GEMM with fused any(sim > phi).
// Block = 256 threads = 8 waves; block tile 256x128; wave tile 32x128
// (2 A-frags x 8 B-frags -> 16 WMMAs per 20 ds_load_b128).
// Async global->LDS staging of 256x32 A and 128x32 B per K-chunk.
// ---------------------------------------------------------------------------
__global__ __launch_bounds__(256) void sim_any_kernel(
    const _Float16* __restrict__ Exh, const _Float16* __restrict__ Eyh,
    float* __restrict__ out)
{
    __shared__ __align__(16) _Float16 smA[2][BLK_M * LDS_STRIDE];  // 2 x 20KB
    __shared__ __align__(16) _Float16 smB[2][BLK_N * LDS_STRIDE];  // 2 x 10KB

    const int t    = threadIdx.x;
    const int lane = t & 31;
    const int wave = t >> 5;          // 0..7 -> 32-row slice of the 256-row block tile
    const int mBlk = blockIdx.x * BLK_M;
    const int nBlk = blockIdx.y * BLK_N;
    const int mBase = mBlk + wave * 32;

    const int halfSel = (lane >> 4) & 1;
    const int lrow    = lane & 15;

    // Staging map: thread t copies 16B segment (t&3) of rows (t>>2)+{0,64,128,192} of A
    // and rows (t>>2)+{0,64} of B.
    const int seg = t & 3;
    const int r0  = t >> 2;

    auto stage = [&](int kc, int b) {
        const size_t kOff = (size_t)kc * KC + seg * 8;
        #pragma unroll
        for (int rr = 0; rr < 4; ++rr) {
            const int r = r0 + rr * 64;
            async_copy_b128(Exh + (size_t)(mBlk + r) * DIM + kOff,
                            &smA[b][r * LDS_STRIDE + seg * 8]);
        }
        #pragma unroll
        for (int rr = 0; rr < 2; ++rr) {
            const int r = r0 + rr * 64;
            async_copy_b128(Eyh + (size_t)(nBlk + r) * DIM + kOff,
                            &smB[b][r * LDS_STRIDE + seg * 8]);
        }
    };

    v8f zero = {};
    v8f acc[2][8];
    #pragma unroll
    for (int i = 0; i < 2; ++i)
        #pragma unroll
        for (int j = 0; j < 8; ++j)
            acc[i][j] = zero;

    stage(0, 0);
    int cur = 0;

    for (int kc = 0; kc < NCHUNK; ++kc) {
        wait_async0();        // my async copies for chunk kc have landed in LDS
        __syncthreads();      // everyone's landed; prior reads of other buffer done

        if (kc + 1 < NCHUNK) stage(kc + 1, cur ^ 1);

        // A fragments (16x32 f16): lanes 0-15 hold K {0..7,16..23}, 16-31 {8..15,24..31}
        v16h aF[2];
        #pragma unroll
        for (int i = 0; i < 2; ++i) {
            const _Float16* p = &smA[cur][(wave * 32 + i * 16 + lrow) * LDS_STRIDE + halfSel * 8];
            union { v16h v; v8h h[2]; } u;
            u.h[0] = *(const v8h*)(p);
            u.h[1] = *(const v8h*)(p + 16);
            aF[i] = u.v;
        }
        // B fragments (32x16 f16): lane = column, lanes 0-15 K {0..7,16..23}, 16-31 {8..15,24..31}
        v16h bF[8];
        #pragma unroll
        for (int j = 0; j < 8; ++j) {
            const _Float16* p = &smB[cur][(j * 16 + lrow) * LDS_STRIDE + halfSel * 8];
            union { v16h v; v8h h[2]; } u;
            u.h[0] = *(const v8h*)(p);
            u.h[1] = *(const v8h*)(p + 16);
            bF[j] = u.v;
        }

        #pragma unroll
        for (int i = 0; i < 2; ++i)
            #pragma unroll
            for (int j = 0; j < 8; ++j)
                acc[i][j] = __builtin_amdgcn_wmma_f32_16x16x32_f16(
                    false, aF[i], false, bF[j],
                    (short)0, acc[i][j], false, false);

        cur ^= 1;
    }

    // ---- fused any(sim > phi) per output row.
    // C/D layout: VGPR g holds row (g + 8*(lane>=16)), col = lane%16.
    #pragma unroll
    for (int i = 0; i < 2; ++i) {
        unsigned hitLo = 0, hitHi = 0;
        #pragma unroll
        for (int j = 0; j < 8; ++j) {
            #pragma unroll
            for (int g = 0; g < 8; ++g) {
                unsigned long long m = __ballot(acc[i][j][g] > PHI);
                unsigned lo = (unsigned)m & 0xFFFFu;
                unsigned hi = ((unsigned)(m >> 16)) & 0xFFFFu;
                if (lo) hitLo |= (1u << g);
                if (hi) hitHi |= (1u << g);
            }
        }
        if (lane == 0) {
            #pragma unroll
            for (int g = 0; g < 8; ++g) {
                if (hitLo & (1u << g)) out[mBase + i * 16 + g]     = 1.0f;
                if (hitHi & (1u << g)) out[mBase + i * 16 + g + 8] = 1.0f;
            }
        }
    }
}

// ---------------------------------------------------------------------------
// Host-side launcher. Workspace layout: [Exh fp16 16MiB][Eyh fp16 16MiB].
// ---------------------------------------------------------------------------
extern "C" void kernel_launch(void* const* d_in, const int* in_sizes, int n_in,
                              void* d_out, int out_size, void* d_ws, size_t ws_size,
                              hipStream_t stream) {
    (void)in_sizes; (void)n_in; (void)out_size; (void)ws_size;

    const float* Ex = (const float*)d_in[0];
    const float* Ey = (const float*)d_in[1];
    float*       out = (float*)d_out;

    _Float16* Exh = (_Float16*)d_ws;
    _Float16* Eyh = Exh + (size_t)N_ROWS * DIM;

    zero_out_kernel<<<(N_ROWS + 255) / 256, 256, 0, stream>>>(out, N_ROWS);

    normalize_f16_kernel<<<2 * N_ROWS, 256, 0, stream>>>(Ex, Ey, Exh, Eyh);

    dim3 grid(N_ROWS / BLK_M, N_ROWS / BLK_N);
    sim_any_kernel<<<grid, 256, 0, stream>>>(Exh, Eyh, out);
}
